// LocalPoolPointnet_70111046140668
// MI455X (gfx1250) — matrix operand order repaired and checked by reading, not compiled, not run on hardware
//
#include <hip/hip_runtime.h>

// ---------------------------------------------------------------------------
// LocalPoolPointnet for MI455X (gfx1250, wave32, WMMA).
// Working set (~115MB net+planes+idx) fits in 192MB L2; all matmuls run as
// v_wmma_f32_16x16x32_f16 (f16 in, f32 acc). Scatter-mean counts are index-
// invariant -> inverse counts computed once, reused for all 5 scatter rounds.
// ---------------------------------------------------------------------------

#define BATCH 8
#define TPTS  100000
#define RESO  128
#define SEG   (RESO * RESO)          // 16384
#define HDIM  32
#define BT    (BATCH * TPTS)         // 800000
#define NPLANE3 (3 * BATCH * SEG)    // 393216

typedef __attribute__((ext_vector_type(16))) _Float16 v16h;
typedef __attribute__((ext_vector_type(8)))  float    v8f;

// ---- WMMA fragment loaders (CDNA5 16-bit layouts, ISA 7.12.2) --------------
// A 16x32: lane l<16 = row l, K in {0..7, 16..23}; lane l+16 same row,
//          K in {8..15, 24..31}; VGPR v holds consecutive pair.
__device__ __forceinline__ v16h ldsA(const _Float16* act, int stride,
                                     int kbase, bool relu) {
  const int lane = threadIdx.x & 31;
  const int row  = lane & 15;
  const int hi   = lane >> 4;
  v16h a;
#pragma unroll
  for (int v = 0; v < 8; ++v) {
    int k = kbase + ((v < 4) ? (2 * v) : (16 + 2 * (v - 4))) + (hi ? 8 : 0);
    _Float16 e0 = act[row * stride + k];
    _Float16 e1 = act[row * stride + k + 1];
    if (relu) {
      e0 = (e0 > (_Float16)0.f) ? e0 : (_Float16)0.f;
      e1 = (e1 > (_Float16)0.f) ? e1 : (_Float16)0.f;
    }
    a[2 * v]     = e0;
    a[2 * v + 1] = e1;
  }
  return a;
}

// B 32x16: lane n<16 = col n, K in {0..7, 16..23}; lane n+16 = col n, K in {8..15, 24..31}.
// Weights stored transposed in LDS as [N][K] so K pairs are contiguous.
__device__ __forceinline__ v16h ldsB(const _Float16* wt, int kdim,
                                     int tile, int kbase) {
  const int lane = threadIdx.x & 31;
  const int col  = lane & 15;
  const int hi   = lane >> 4;
  v16h b;
#pragma unroll
  for (int v = 0; v < 8; ++v) {
    int k = kbase + ((v < 4) ? (2 * v) : (16 + 2 * (v - 4))) + (hi ? 8 : 0);
    b[2 * v]     = wt[(tile * 16 + col) * kdim + k];
    b[2 * v + 1] = wt[(tile * 16 + col) * kdim + k + 1];
  }
  return b;
}

// ResnetBlockFC on a 16-point tile: out = x@Wsc + fc1(relu(fc0(relu(x)))).
// act: [16][64] f16 (LDS, per-wave). mid: [16][32] f16 scratch (LDS, per-wave).
__device__ __forceinline__ void resblock16(_Float16* act, _Float16* mid,
    const _Float16* w0t, const float* b0, const _Float16* w1t, const float* b1,
    const _Float16* wsct, v8f* dout) {
  const int lane = threadIdx.x & 31;
  const int n    = lane & 15;
  const int hi   = lane >> 4;
  // fc0: relu(x) @ W0 + b0   (K = 64 -> two WMMA per N-tile)
#pragma unroll
  for (int t = 0; t < 2; ++t) {
    v8f c;
    float bb = b0[t * 16 + n];
#pragma unroll
    for (int j = 0; j < 8; ++j) c[j] = bb;
    c = __builtin_amdgcn_wmma_f32_16x16x32_f16(false, ldsA(act, 64, 0, true),
        false, ldsB(w0t, 64, t, 0), (short)0, c, false, false);
    c = __builtin_amdgcn_wmma_f32_16x16x32_f16(false, ldsA(act, 64, 32, true),
        false, ldsB(w0t, 64, t, 32), (short)0, c, false, false);
#pragma unroll
    for (int r = 0; r < 8; ++r) {           // relu + transpose via LDS
      float v = c[r] > 0.f ? c[r] : 0.f;
      mid[(r + hi * 8) * 32 + t * 16 + n] = (_Float16)v;
    }
  }
  // fc1(mid) + b1, then chain shortcut x@Wsc into the same accumulator
#pragma unroll
  for (int t = 0; t < 2; ++t) {
    v8f c;
    float bb = b1[t * 16 + n];
#pragma unroll
    for (int j = 0; j < 8; ++j) c[j] = bb;
    c = __builtin_amdgcn_wmma_f32_16x16x32_f16(false, ldsA(mid, 32, 0, false),
        false, ldsB(w1t, 32, t, 0), (short)0, c, false, false);
    c = __builtin_amdgcn_wmma_f32_16x16x32_f16(false, ldsA(act, 64, 0, false),
        false, ldsB(wsct, 64, t, 0), (short)0, c, false, false);
    c = __builtin_amdgcn_wmma_f32_16x16x32_f16(false, ldsA(act, 64, 32, false),
        false, ldsB(wsct, 64, t, 32), (short)0, c, false, false);
    dout[t] = c;
  }
}

// ---- helpers ---------------------------------------------------------------
__device__ __forceinline__ int cellOf(float a, float b) {
  const float inv = 1.0f / (1.0f + 0.1f + 1e-5f);
  float xa = fminf(fmaxf(a * inv + 0.5f, 0.f), 1.f - 1e-5f);
  float xb = fminf(fmaxf(b * inv + 0.5f, 0.f), 1.f - 1e-5f);
  int ga = (int)(xa * (float)RESO);
  int gb = (int)(xb * (float)RESO);
  return ga + RESO * gb;
}

__global__ __launch_bounds__(256) void k_zero(unsigned* ptr, long n) {
  long i = (long)blockIdx.x * 256 + threadIdx.x;
  if (i < n) ptr[i] = 0u;
}

__global__ __launch_bounds__(256) void k_idx_count(const float* __restrict__ p,
    int* __restrict__ idxw, unsigned* __restrict__ cnt) {
  long i = (long)blockIdx.x * 256 + threadIdx.x;
  if (i >= BT) return;
  float x = p[i * 3 + 0], y = p[i * 3 + 1], z = p[i * 3 + 2];
  int b = (int)(i / TPTS);
  int c0 = cellOf(x, z);   // plane (0,2)
  int c1 = cellOf(x, y);   // plane (0,1)
  int c2 = cellOf(y, z);   // plane (1,2)
  idxw[0L * BT + i] = c0;
  idxw[1L * BT + i] = c1;
  idxw[2L * BT + i] = c2;
  atomicAdd(&cnt[(0 * BATCH + b) * SEG + c0], 1u);
  atomicAdd(&cnt[(1 * BATCH + b) * SEG + c1], 1u);
  atomicAdd(&cnt[(2 * BATCH + b) * SEG + c2], 1u);
}

__global__ __launch_bounds__(256) void k_invcnt(const unsigned* __restrict__ cnt,
                                                float* __restrict__ invc) {
  long i = (long)blockIdx.x * 256 + threadIdx.x;
  if (i < NPLANE3) invc[i] = 1.0f / fmaxf((float)cnt[i], 1.0f);
}

// ---- posFC (3->64) + block0, WMMA ------------------------------------------
__global__ __launch_bounds__(256) void k_pos_block0(const float* __restrict__ p,
    const float* __restrict__ Wpos, const float* __restrict__ bpos,
    const float* __restrict__ W0, const float* __restrict__ b0,
    const float* __restrict__ W1, const float* __restrict__ b1,
    const float* __restrict__ Wsc, _Float16* __restrict__ net) {
  __shared__ float    s_wpos[192], s_bpos[64], s_b0[32], s_b1[32];
  __shared__ _Float16 s_w0t[32 * 64], s_w1t[32 * 32], s_wsct[32 * 64];
  __shared__ _Float16 s_act[8 * 16 * 64], s_mid[8 * 16 * 32];
  const int tid = threadIdx.x;
  for (int i = tid; i < 192; i += 256) s_wpos[i] = Wpos[i];
  for (int i = tid; i < 64;  i += 256) s_bpos[i] = bpos[i];
  for (int i = tid; i < 2048; i += 256) {
    int k = i >> 5, n = i & 31;
    s_w0t[n * 64 + k]  = (_Float16)W0[i];
    s_wsct[n * 64 + k] = (_Float16)Wsc[i];
  }
  for (int i = tid; i < 1024; i += 256) {
    int k = i >> 5, n = i & 31;
    s_w1t[n * 32 + k] = (_Float16)W1[i];
  }
  if (tid < 32) { s_b0[tid] = b0[tid]; s_b1[tid] = b1[tid]; }
  __syncthreads();

  const int wave = tid >> 5, lane = tid & 31;
  const long pt0 = ((long)blockIdx.x * 8 + wave) * 16;
  const int  b   = (int)(pt0 / TPTS);
  const int  t0  = (int)(pt0 % TPTS);
  _Float16* act = s_act + wave * 16 * 64;
  _Float16* mid = s_mid + wave * 16 * 32;

#pragma unroll 4
  for (int j = 0; j < 16; ++j) {
    long pi = (long)b * TPTS + t0 + j;
    float px = p[pi * 3 + 0], py = p[pi * 3 + 1], pz = p[pi * 3 + 2];
    for (int f = lane; f < 64; f += 32) {
      float v = px * s_wpos[f] + py * s_wpos[64 + f] + pz * s_wpos[128 + f]
              + s_bpos[f];
      act[j * 64 + f] = (_Float16)v;
    }
  }
  v8f d[2];
  resblock16(act, mid, s_w0t, s_b0, s_w1t, s_b1, s_wsct, d);
  const int n = lane & 15, hi = lane >> 4;
#pragma unroll
  for (int r = 0; r < 8; ++r) {
    long m = (long)b * TPTS + t0 + r + hi * 8;
    net[m * 32 + n]      = (_Float16)d[0][r];
    net[m * 32 + n + 16] = (_Float16)d[1][r];
  }
}

// ---- scatter net -> plane accumulators (f32 atomics into L2) ---------------
__global__ __launch_bounds__(256) void k_scatter(const _Float16* __restrict__ net,
    const int* __restrict__ idxw, float* __restrict__ acc) {
  long i = (long)blockIdx.x * 256 + threadIdx.x;
  if (i >= BT) return;
  int b = (int)(i / TPTS);
  float v[HDIM];
#pragma unroll
  for (int f = 0; f < HDIM; ++f) v[f] = (float)net[i * HDIM + f];
#pragma unroll
  for (int pl = 0; pl < 3; ++pl) {
    int cell = idxw[(long)pl * BT + i];
    float* dst = acc + ((long)(pl * BATCH + b) * SEG + cell) * HDIM;
#pragma unroll
    for (int f = 0; f < HDIM; ++f) atomicAdd(dst + f, v[f]);
  }
}

// ---- pool_local gather + resblock(i), WMMA ---------------------------------
__global__ __launch_bounds__(256) void k_gather_resblock(
    _Float16* __restrict__ net, const int* __restrict__ idxw,
    const float* __restrict__ acc, const float* __restrict__ invc,
    const float* __restrict__ W0, const float* __restrict__ b0,
    const float* __restrict__ W1, const float* __restrict__ b1,
    const float* __restrict__ Wsc) {
  __shared__ float    s_b0[32], s_b1[32];
  __shared__ _Float16 s_w0t[32 * 64], s_w1t[32 * 32], s_wsct[32 * 64];
  __shared__ _Float16 s_act[8 * 16 * 64], s_mid[8 * 16 * 32];
  const int tid = threadIdx.x;
  for (int i = tid; i < 2048; i += 256) {
    int k = i >> 5, n = i & 31;
    s_w0t[n * 64 + k]  = (_Float16)W0[i];
    s_wsct[n * 64 + k] = (_Float16)Wsc[i];
  }
  for (int i = tid; i < 1024; i += 256) {
    int k = i >> 5, n = i & 31;
    s_w1t[n * 32 + k] = (_Float16)W1[i];
  }
  if (tid < 32) { s_b0[tid] = b0[tid]; s_b1[tid] = b1[tid]; }
  __syncthreads();

  const int wave = tid >> 5, lane = tid & 31;
  const long pt0 = ((long)blockIdx.x * 8 + wave) * 16;
  const int  b   = (int)(pt0 / TPTS);
  const int  t0  = (int)(pt0 % TPTS);
  _Float16* act = s_act + wave * 16 * 64;
  _Float16* mid = s_mid + wave * 16 * 32;

  // lane f gathers feature f: x -> act[:, 0:32], pooled -> act[:, 32:64]
  for (int j = 0; j < 16; ++j) {
    long pi = (long)b * TPTS + t0 + j;
    act[j * 64 + lane] = net[pi * 32 + lane];
    float pool = 0.f;
#pragma unroll
    for (int pl = 0; pl < 3; ++pl) {
      int  cell = idxw[(long)pl * BT + pi];
      long base = (long)(pl * BATCH + b) * SEG + cell;
      pool += acc[base * 32 + lane] * invc[base];
    }
    act[j * 64 + 32 + lane] = (_Float16)pool;
  }
  v8f d[2];
  resblock16(act, mid, s_w0t, s_b0, s_w1t, s_b1, s_wsct, d);
  const int n = lane & 15, hi = lane >> 4;
#pragma unroll
  for (int r = 0; r < 8; ++r) {
    long m = (long)b * TPTS + t0 + r + hi * 8;
    net[m * 32 + n]      = (_Float16)d[0][r];
    net[m * 32 + n + 16] = (_Float16)d[1][r];
  }
}

// ---- final FC (32->32) WMMA + scatter into output planes -------------------
__global__ __launch_bounds__(256) void k_final(const _Float16* __restrict__ net,
    const int* __restrict__ idxw, const float* __restrict__ Wc,
    const float* __restrict__ bc, float* __restrict__ out) {
  __shared__ float    s_bc[32];
  __shared__ _Float16 s_wct[32 * 32];
  __shared__ _Float16 s_act[8 * 16 * 32];
  const int tid = threadIdx.x;
  for (int i = tid; i < 1024; i += 256) {
    int k = i >> 5, n = i & 31;
    s_wct[n * 32 + k] = (_Float16)Wc[i];
  }
  if (tid < 32) s_bc[tid] = bc[tid];
  __syncthreads();

  const int wave = tid >> 5, lane = tid & 31;
  const long pt0 = ((long)blockIdx.x * 8 + wave) * 16;
  const int  b   = (int)(pt0 / TPTS);
  const int  t0  = (int)(pt0 % TPTS);
  _Float16* act = s_act + wave * 16 * 32;
  for (int j = 0; j < 16; ++j)
    act[j * 32 + lane] = net[((long)b * TPTS + t0 + j) * 32 + lane];

  const int n = lane & 15, hi = lane >> 4;
  v8f c[2];
#pragma unroll
  for (int t = 0; t < 2; ++t) {
    float bb = s_bc[t * 16 + n];
#pragma unroll
    for (int j = 0; j < 8; ++j) c[t][j] = bb;
    c[t] = __builtin_amdgcn_wmma_f32_16x16x32_f16(false,
        ldsA(act, 32, 0, false), false, ldsB(s_wct, 32, t, 0),
        (short)0, c[t], false, false);
  }
#pragma unroll
  for (int r = 0; r < 8; ++r) {
    long m = (long)b * TPTS + t0 + r + hi * 8;
#pragma unroll
    for (int pl = 0; pl < 3; ++pl) {
      int  cell = idxw[(long)pl * BT + m];
      long pb   = (long)(pl * BATCH + b) * 32;
      atomicAdd(&out[(pb + n) * SEG + cell],        c[0][r]);
      atomicAdd(&out[(pb + n + 16) * SEG + cell],   c[1][r]);
    }
  }
}

__global__ __launch_bounds__(256) void k_scale_out(float* __restrict__ out,
    const float* __restrict__ invc) {
  long i = (long)blockIdx.x * 256 + threadIdx.x;
  long total = 3L * BATCH * 32 * SEG;
  if (i >= total) return;
  long s  = i % SEG;
  long pb = (i / SEG) / 32;           // pl*BATCH + b
  out[i] *= invc[pb * SEG + s];
}

// ---------------------------------------------------------------------------
extern "C" void kernel_launch(void* const* d_in, const int* in_sizes, int n_in,
                              void* d_out, int out_size, void* d_ws, size_t ws_size,
                              hipStream_t stream) {
  const float* p    = (const float*)d_in[0];
  const float* Wpos = (const float*)d_in[1];
  const float* bpos = (const float*)d_in[2];
  const float* W0   = (const float*)d_in[3];
  const float* b0   = (const float*)d_in[4];
  const float* W1   = (const float*)d_in[5];
  const float* b1   = (const float*)d_in[6];
  const float* Wsc  = (const float*)d_in[7];
  const float* Wc   = (const float*)d_in[8];
  const float* bc   = (const float*)d_in[9];
  float* out = (float*)d_out;

  // workspace carve-up (~115 MB)
  char* w = (char*)d_ws;
  int* idxw = (int*)w;                 w += sizeof(int) * 3L * BT;
  unsigned* cnt = (unsigned*)w;        w += sizeof(unsigned) * (long)NPLANE3;
  float* invc = (float*)w;             w += sizeof(float) * (long)NPLANE3;
  float* acc = (float*)w;              w += sizeof(float) * (long)NPLANE3 * 32;
  _Float16* net = (_Float16*)w;

  const long accN = (long)NPLANE3 * 32;            // 12.58M f32
  const long outN = 3L * BATCH * 32 * SEG;         // 12.58M f32
  const int  ptBlocks   = (BT + 255) / 256;        // 3125
  const int  tileBlocks = BT / (16 * 8);           // 6250 (exact)

  // counts + inverse counts (index-only, computed once per call)
  k_zero<<<(NPLANE3 + 255) / 256, 256, 0, stream>>>(cnt, NPLANE3);
  k_idx_count<<<ptBlocks, 256, 0, stream>>>(p, idxw, cnt);
  k_invcnt<<<(NPLANE3 + 255) / 256, 256, 0, stream>>>(cnt, invc);

  // posFC + resblock 0
  k_pos_block0<<<tileBlocks, 256, 0, stream>>>(p, Wpos, bpos,
      W0, b0, W1, b1, Wsc, net);

  // 4 rounds of pool_local + resblock i
  for (int i = 1; i < 5; ++i) {
    k_zero<<<(int)((accN + 255) / 256), 256, 0, stream>>>((unsigned*)acc, accN);
    k_scatter<<<ptBlocks, 256, 0, stream>>>(net, idxw, acc);
    k_gather_resblock<<<tileBlocks, 256, 0, stream>>>(net, idxw, acc, invc,
        W0 + i * 64 * 32, b0 + i * 32, W1 + i * 32 * 32, b1 + i * 32,
        Wsc + i * 64 * 32);
  }

  // c = net @ Wc + bc, scatter-mean into output planes
  k_zero<<<(int)((outN + 255) / 256), 256, 0, stream>>>((unsigned*)out, outN);
  k_final<<<tileBlocks, 256, 0, stream>>>(net, idxw, Wc, bc, out);
  k_scale_out<<<(int)((outN + 255) / 256), 256, 0, stream>>>(out, invc);
}
